// MultiHeadSelfAttention_40080634806734
// MI455X (gfx1250) — matrix operand order repaired and checked by reading, not compiled, not run on hardware
//
#include <hip/hip_runtime.h>

typedef __attribute__((ext_vector_type(16))) _Float16 v16h;
typedef __attribute__((ext_vector_type(8)))  _Float16 v8h;
typedef __attribute__((ext_vector_type(4)))  _Float16 v4h;
typedef __attribute__((ext_vector_type(8)))  float    v8f;
typedef __attribute__((ext_vector_type(4)))  unsigned int u32x4;
typedef __attribute__((ext_vector_type(8)))  int      i32x8;
typedef __attribute__((ext_vector_type(4)))  int      i32x4;

#define SEQ   2048
#define BSZ   2
#define EMB   512
#define HEADS 8
#define HD    64
#define BH    (BSZ * HEADS)   // 16
#define SHIFT 8.0f            // fixed softmax shift (shift-invariant, overflow-safe)

static __device__ __forceinline__ v8f wmma_f16(v16h a, v16h b, v8f c) {
    // D = A(16x32 f16) * B(32x16 f16) + C(16x16 f32)
    return __builtin_amdgcn_wmma_f32_16x16x32_f16(
        false, a, false, b, (short)0, c, false, false);
}

// A fragment: K(i) = i + 16*hi over a K-contiguous row -> 1x 32B LDS load.
static __device__ __forceinline__ v16h load_afrag(const _Float16* row, int hi) {
    return *(const v16h*)(row + hi * 16);
}
// B fragment: K(i) = i + 16*hi over a K-contiguous row -> 1x 32B LDS load.
static __device__ __forceinline__ v16h load_bfrag(const _Float16* col, int hi) {
    return *(const v16h*)(col + hi * 16);
}

// TDM: 1-D copy of a contiguous 8 KB tile (4096 f16) from global to LDS.
// D# per CDNA5 ISA: group0={count=1, lds_addr,
// global_addr, type=2}; group1={data_size=1(2B), tensor_dim0=tile_dim0=4096,
// tile_dim1=0 -> 1-D}; groups 2/3 zero (<=2-D). Tracked by TENSORcnt.
// 6-arg builtin form (clang-23 / therock-10.0 headers).
static __device__ __forceinline__ void tdm_copy_8k(const _Float16* gsrc, _Float16* ldst) {
    unsigned long long ga = (unsigned long long)(uintptr_t)gsrc;
    u32x4 g0;
    g0[0] = 1u;                                             // count=1
    g0[1] = (unsigned int)(uintptr_t)ldst;                  // lds byte address
    g0[2] = (unsigned int)ga;                               // global_addr[31:0]
    g0[3] = (unsigned int)((ga >> 32) & 0x01FFFFFFu) | 0x80000000u; // [56:32]|type=2
    i32x8 g1 = {};
    g1[0] = 0x00010000;                                     // data_size=1 (2B), mask=0
    g1[1] = (int)(4096u << 16);                             // tensor_dim0[15:0]=4096
    g1[2] = 0;                                              // tensor_dim0 hi / dim1 lo
    g1[3] = (int)(4096u << 16);                             // tile_dim0=4096
    g1[4] = 0;                                              // tile_dim1=0 (1-D)
    g1[5] = 4096;                                           // tensor_dim0_stride
    g1[6] = 0; g1[7] = 0;
    i32x4 z4 = {};
    i32x8 z8 = {};
    __builtin_amdgcn_tensor_load_to_lds(g0, g1, z4, z4, z8, 0);
}

// ---------------------------------------------------------------------------
// Kernel 1: fused QKV projection.  M=4096 (r=s*2+b), N=1536, K=512.
// Epilogue scatters to head-major f16 Q (pre-scaled 1/8), K, V buffers.
// ---------------------------------------------------------------------------
__global__ __launch_bounds__(512)
void qkv_proj_kernel(const float* __restrict__ x, const float* __restrict__ W,
                     const float* __restrict__ bias,
                     _Float16* __restrict__ qbuf, _Float16* __restrict__ kbuf,
                     _Float16* __restrict__ vbuf)
{
    __shared__ __align__(32) _Float16 As[64][32];   // [row][k]
    __shared__ __align__(32) _Float16 Bs[64][32];   // [col][k]  (K-contiguous)
    const int tid  = threadIdx.x;
    const int lane = tid & 31, w = tid >> 5;
    const int hi   = lane >> 4, ncol = lane & 15, mrow = lane & 15;
    const int mi   = w >> 2,  ni = w & 3;
    const int Mb   = blockIdx.x * 64, Nb = blockIdx.y * 64;
    const int sr   = tid >> 3, sc4 = (tid & 7) * 4;

    v8f acc = {};
    for (int kk = 0; kk < 512; kk += 32) {
        float4 xa = *(const float4*)&x[(size_t)(Mb + sr) * 512 + kk + sc4];
        float4 wa = *(const float4*)&W[(size_t)(Nb + sr) * 512 + kk + sc4];
        As[sr][sc4+0] = (_Float16)xa.x; As[sr][sc4+1] = (_Float16)xa.y;
        As[sr][sc4+2] = (_Float16)xa.z; As[sr][sc4+3] = (_Float16)xa.w;
        Bs[sr][sc4+0] = (_Float16)wa.x; Bs[sr][sc4+1] = (_Float16)wa.y;
        Bs[sr][sc4+2] = (_Float16)wa.z; Bs[sr][sc4+3] = (_Float16)wa.w;
        __syncthreads();
        v16h a = load_afrag(&As[mi * 16 + mrow][0], hi);
        v16h b = load_bfrag(&Bs[ni * 16 + ncol][0], hi);
        acc = wmma_f16(a, b, acc);
        __syncthreads();
    }
    const int   col = Nb + ni * 16 + ncol;
    const float bsv = bias[col];
    const int   sec = col >> 9, f = col & 511, hh = f >> 6, d = f & 63;
    _Float16*   dbuf = (sec == 0) ? qbuf : ((sec == 1) ? kbuf : vbuf);
    const float scl  = (sec == 0) ? 0.125f : 1.0f;     // 1/sqrt(64) on Q
    #pragma unroll
    for (int j = 0; j < 8; ++j) {
        int r = Mb + mi * 16 + j + hi * 8;
        int dst = ((r & 1) * HEADS + hh) * (SEQ * HD) + (r >> 1) * HD + d;
        dbuf[dst] = (_Float16)((acc[j] + bsv) * scl);
    }
}

// ---------------------------------------------------------------------------
// Kernel 2: attention per (b, 64-row q-tile), looping heads.
// 256 KB Pbig in LDS (one exp/element, single score pass).  Q and K tiles are
// streamed by the Tensor Data Mover (double-buffered K: DMA(i+1) overlaps
// compute(i), retired with s_wait_tensorcnt).
// ---------------------------------------------------------------------------
__global__ __launch_bounds__(512)
void attn_kernel(const _Float16* __restrict__ qbuf, const _Float16* __restrict__ kbuf,
                 const _Float16* __restrict__ vbuf, _Float16* __restrict__ attnbuf,
                 float* __restrict__ aw)
{
    __shared__ __align__(32) _Float16 Pbig[64][SEQ];  // 256 KB unnormalized probs
    __shared__ __align__(32) _Float16 Qs[64][64];     // [q][d]
    __shared__ __align__(32) _Float16 Ks[2][64][64];  // double-buffered K tiles
    __shared__ __align__(32) _Float16 Vt[64][64];     // [d][k] (B-frag ready)
    __shared__ float partl[4][64];
    __shared__ float rowinv[64];

    const int tid  = threadIdx.x, lane = tid & 31, w = tid >> 5;
    const int hi   = lane >> 4, ncol = lane & 15, mrow = lane & 15;
    const int qi   = w >> 2, kdi = w & 3;          // wave tile role
    const int b    = blockIdx.x >> 5;
    const int qbase = (blockIdx.x & 31) * 64;
    const int sr   = tid >> 3, sc8 = (tid & 7) * 8;  // staging: 512 thr = 64x8

    for (int h = 0; h < HEADS; ++h) {
        const size_t hoff = (size_t)(b * HEADS + h) * SEQ * HD;
        float l8[8];
        #pragma unroll
        for (int j = 0; j < 8; ++j) l8[j] = 0.f;
        __syncthreads();   // previous head fully done with Pbig/Qs/Ks
        if (w == 0) {      // one wave drives the TDM; EXEC-independent DMA
            tdm_copy_8k(&qbuf[hoff + (size_t)qbase * HD], &Qs[0][0]);
            tdm_copy_8k(&kbuf[hoff], &Ks[0][0][0]);
        }

        // ---- pass A: scores once, exp once, fill Pbig, accumulate row sums
        #define SCORE_STEP(i)                                                   \
        {                                                                       \
            const _Float16 (*Kc)[64] = Ks[(i) & 1];                             \
            v8f c = {};                                                         \
            _Pragma("unroll")                                                   \
            for (int ch = 0; ch < 2; ++ch) {                                    \
                v16h a  = load_afrag(&Qs[qi * 16 + mrow][ch * 32], hi);         \
                v16h bb = load_bfrag(&Kc[kdi * 16 + ncol][ch * 32], hi);        \
                c = wmma_f16(a, bb, c);                                         \
            }                                                                   \
            _Pragma("unroll")                                                   \
            for (int j = 0; j < 8; ++j) {                                       \
                int   row = qi * 16 + j + hi * 8;                               \
                float p   = __expf(c[j] - SHIFT);                               \
                l8[j] += p;                                                     \
                Pbig[row][(i) * 64 + kdi * 16 + ncol] = (_Float16)p;            \
            }                                                                   \
        }
        for (int i = 0; i < 31; ++i) {
            if (w == 0)    // prefetch next K tile; overlaps with compute below
                tdm_copy_8k(&kbuf[hoff + (size_t)(i + 1) * 64 * HD],
                            &Ks[(i + 1) & 1][0][0]);
            __builtin_amdgcn_s_wait_tensorcnt(1);   // tile i (and Q) landed
            __syncthreads();
            SCORE_STEP(i);
            __syncthreads();
        }
        __builtin_amdgcn_s_wait_tensorcnt(0);       // peeled last tile
        __syncthreads();
        SCORE_STEP(31);
        #undef SCORE_STEP

        // plain add-reduction of row sums (16 lanes per half share a row set)
        #pragma unroll
        for (int j = 0; j < 8; ++j) {
            float l = l8[j];
            #pragma unroll
            for (int mask = 1; mask < 16; mask <<= 1) l += __shfl_xor(l, mask, 32);
            if (ncol == 0) partl[kdi][qi * 16 + j + hi * 8] = l;
        }
        __syncthreads();
        if (tid < 64)
            rowinv[tid] = 1.0f / (partl[0][tid] + partl[1][tid] +
                                  partl[2][tid] + partl[3][tid]);
        __syncthreads();

        // ---- pass B: O_un = P_un x V  (A-fragments straight from Pbig)
        v8f o = {};
        for (int kb = 0; kb < SEQ; kb += 64) {
            v8h vv = *(const v8h*)&vbuf[hoff + (size_t)(kb + sr) * HD + sc8];
            #pragma unroll
            for (int u = 0; u < 8; ++u) Vt[sc8 + u][sr] = vv[u];  // transpose V
            __syncthreads();
            #pragma unroll
            for (int ch = 0; ch < 2; ++ch) {
                v16h a  = load_afrag(&Pbig[qi * 16 + mrow][kb + ch * 32], hi);
                v16h bb = load_bfrag(&Vt[kdi * 16 + ncol][ch * 32], hi);
                o = wmma_f16(a, bb, o);
            }
            __syncthreads();
        }
        // write normalized attn rows as [s][b][h*64 + d] f16
        #pragma unroll
        for (int j = 0; j < 8; ++j) {
            int row = qi * 16 + j + hi * 8;
            int s   = qbase + row;
            attnbuf[(size_t)(s * BSZ + b) * EMB + h * HD + kdi * 16 + ncol] =
                (_Float16)(o[j] * rowinv[row]);
        }

        // ---- attn_weights: vectorized sweep of Pbig, float4 global RMW
        {
            const size_t awrow = ((size_t)b * SEQ + (qbase + sr)) * SEQ;
            const float  scl   = rowinv[sr] * 0.125f;   // mean over 8 heads
            for (int kb = 0; kb < SEQ; kb += 64) {
                v8h p8 = *(const v8h*)&Pbig[sr][kb + sc8];
                float4 f0, f1;
                f0.x = (float)p8[0] * scl; f0.y = (float)p8[1] * scl;
                f0.z = (float)p8[2] * scl; f0.w = (float)p8[3] * scl;
                f1.x = (float)p8[4] * scl; f1.y = (float)p8[5] * scl;
                f1.z = (float)p8[6] * scl; f1.w = (float)p8[7] * scl;
                float4* g = (float4*)&aw[awrow + kb + sc8];
                if (h != 0) {                      // single-writer RMW, deterministic
                    float4 p0 = g[0], p1 = g[1];
                    f0.x += p0.x; f0.y += p0.y; f0.z += p0.z; f0.w += p0.w;
                    f1.x += p1.x; f1.y += p1.y; f1.z += p1.z; f1.w += p1.w;
                }
                g[0] = f0; g[1] = f1;
            }
        }
    }
}

// ---------------------------------------------------------------------------
// Kernel 3: output projection.  M=4096, N=512, K=512; A already f16.
// ---------------------------------------------------------------------------
__global__ __launch_bounds__(512)
void out_proj_kernel(const _Float16* __restrict__ attnbuf, const float* __restrict__ W,
                     const float* __restrict__ bias, float* __restrict__ out)
{
    __shared__ __align__(32) _Float16 As[64][32];   // [row][k]
    __shared__ __align__(32) _Float16 Bs[64][32];   // [col][k]
    const int tid  = threadIdx.x, lane = tid & 31, w = tid >> 5;
    const int hi   = lane >> 4, ncol = lane & 15, mrow = lane & 15;
    const int mi   = w >> 2, ni = w & 3;
    const int Mb   = blockIdx.x * 64, Nb = blockIdx.y * 64;
    const int sr   = tid >> 3, sc4 = (tid & 7) * 4;

    v8f acc = {};
    for (int kk = 0; kk < 512; kk += 32) {
        *(v4h*)&As[sr][sc4] = *(const v4h*)&attnbuf[(size_t)(Mb + sr) * 512 + kk + sc4];
        float4 wa = *(const float4*)&W[(size_t)(Nb + sr) * 512 + kk + sc4];
        Bs[sr][sc4+0] = (_Float16)wa.x; Bs[sr][sc4+1] = (_Float16)wa.y;
        Bs[sr][sc4+2] = (_Float16)wa.z; Bs[sr][sc4+3] = (_Float16)wa.w;
        __syncthreads();
        v16h a = load_afrag(&As[mi * 16 + mrow][0], hi);
        v16h b = load_bfrag(&Bs[ni * 16 + ncol][0], hi);
        acc = wmma_f16(a, b, acc);
        __syncthreads();
    }
    const int   col = Nb + ni * 16 + ncol;
    const float bsv = bias[col];
    #pragma unroll
    for (int j = 0; j < 8; ++j) {
        int r = Mb + mi * 16 + j + hi * 8;
        out[(size_t)r * 512 + col] = acc[j] + bsv;
    }
}

// ---------------------------------------------------------------------------
extern "C" void kernel_launch(void* const* d_in, const int* in_sizes, int n_in,
                              void* d_out, int out_size, void* d_ws, size_t ws_size,
                              hipStream_t stream)
{
    const float* x    = (const float*)d_in[0];
    const float* Win  = (const float*)d_in[1];
    const float* bin  = (const float*)d_in[2];
    const float* Wout = (const float*)d_in[3];
    const float* bout = (const float*)d_in[4];

    float* out = (float*)d_out;                         // [S,B,E] = 2,097,152 f32
    float* aw  = out + (size_t)SEQ * BSZ * EMB;         // [B,S,S] = 8,388,608 f32

    // workspace: head-major f16 Q/K/V (4 MB each) + pre-projection attn (4 MB)
    _Float16* qbuf    = (_Float16*)d_ws;
    _Float16* kbuf    = qbuf + (size_t)BH * SEQ * HD;
    _Float16* vbuf    = kbuf + (size_t)BH * SEQ * HD;
    _Float16* attnbuf = vbuf + (size_t)BH * SEQ * HD;

    qkv_proj_kernel<<<dim3(64, 24), 512, 0, stream>>>(x, Win, bin, qbuf, kbuf, vbuf);
    attn_kernel    <<<dim3(64),     512, 0, stream>>>(qbuf, kbuf, vbuf, attnbuf, aw);
    out_proj_kernel<<<dim3(64, 8),  512, 0, stream>>>(attnbuf, Wout, bout, out);
}
